// FeatureFusion_72000831750186
// MI455X (gfx1250) — matrix operand order repaired and checked by reading, not compiled, not run on hardware
//
#include <hip/hip_runtime.h>
#include <cstdint>
#include <cstddef>

// ---------------- problem constants ----------------
#define BATCH  128
#define CDIM   512
#define HWNN   576
#define KSEL   16
#define NTXT   128
#define NHEAD  8
#define DHEAD  64
#define NLAYER 4
#define FDIM   2048
#define TQ     (BATCH * KSEL)   // 2048 query tokens
#define TKV    (BATCH * HWNN)   // 73728 kv tokens
#define EPSLN  1e-5f

typedef __attribute__((ext_vector_type(16))) __bf16 v16bf;
typedef __attribute__((ext_vector_type(8)))  float  v8f;

union FragBF { v16bf v; unsigned int u[8]; };

__device__ __forceinline__ unsigned short f2bf(float x) {
  unsigned int u = __float_as_uint(x);
  unsigned int r = u + 0x7FFFu + ((u >> 16) & 1u);   // round-to-nearest-even
  return (unsigned short)(r >> 16);
}
__device__ __forceinline__ unsigned int pack2bf(float a, float b) {
  return (unsigned int)f2bf(a) | ((unsigned int)f2bf(b) << 16);
}
__device__ __forceinline__ float quick_gelu(float x) {
  // x * sigmoid(1.702 x) with fast hardware reciprocal (no IEEE div chain)
  return x * __builtin_amdgcn_rcpf(1.f + __expf(-1.702f * x));
}

// ---------------- WMMA fragment loaders (contiguous k order k = 16*(lane/16) + s, same for A and B) ----------------
// A: 16x32 MxK bf16. lane&15 = M row; lane>>4 selects K sub-half.
// VGPR v: k0 = kbase + (lane/16)*16 + v*2 ; holds A[m][k0], A[m][k0+1].
// v0..v7 are 16 consecutive bf16 -> compiler emits 2x b128 loads.
__device__ __forceinline__ void loadA_bf(FragBF& f, const unsigned short* A, int lda,
                                         long row0, int kbase, int lane) {
  int m = lane & 15, kh = lane >> 4;
  const unsigned short* base = A + (size_t)(row0 + m) * lda;
#pragma unroll
  for (int v = 0; v < 8; ++v) {
    int k0 = kbase + (kh << 4) + (v << 1);
    f.u[v] = *(const unsigned int*)(base + k0);
  }
}
// B: 32x16 KxN. lane&15 = n; lane>>4 = kh; VGPR v -> k0 = kbase + kh*16 + 2v.
// "nk": memory is B^T row-major, i.e. M[n][k] (weight rows, K^T).
__device__ __forceinline__ void loadB_nk_bf(FragBF& f, const unsigned short* M, int ldm,
                                            long n0, int kbase, int lane) {
  int n = lane & 15, kh = lane >> 4;
  const unsigned short* r = M + (size_t)(n0 + n) * ldm;
#pragma unroll
  for (int v = 0; v < 8; ++v) {
    int k0 = kbase + (kh << 4) + (v << 1);
    f.u[v] = *(const unsigned int*)(r + k0);
  }
}
// "kn": memory is M[k][n] row-major over k (V matrix, img_vis columns).
__device__ __forceinline__ void loadB_kn_bf(FragBF& f, const unsigned short* M, int ldm,
                                            long kbase, long n0, int lane) {
  int n = lane & 15, kh = lane >> 4;
#pragma unroll
  for (int v = 0; v < 8; ++v) {
    long k0 = kbase + (kh << 4) + (v << 1);
    unsigned int lo = M[(size_t)k0 * ldm + n0 + n];
    unsigned int hi = M[(size_t)(k0 + 1) * ldm + n0 + n];
    f.u[v] = lo | (hi << 16);
  }
}
__device__ __forceinline__ void loadB_kn_f32(FragBF& f, const float* M, int ldm,
                                             long kbase, long n0, int lane) {
  int n = lane & 15, kh = lane >> 4;
#pragma unroll
  for (int v = 0; v < 8; ++v) {
    long k0 = kbase + (kh << 4) + (v << 1);
    f.u[v] = pack2bf(M[(size_t)k0 * ldm + n0 + n], M[(size_t)(k0 + 1) * ldm + n0 + n]);
  }
}

__device__ __forceinline__ v8f wmma_bf16(const FragBF& a, const FragBF& b, v8f c) {
  return __builtin_amdgcn_wmma_f32_16x16x32_bf16(false, a.v, false, b.v, (short)0, c,
                                                 false, false);
}

// ================= weight pre-conversion =================
__global__ void cvt_bf16_kernel(const float* __restrict__ src,
                                unsigned short* __restrict__ dst, size_t n) {
  for (size_t i = (size_t)blockIdx.x * blockDim.x + threadIdx.x; i < n;
       i += (size_t)gridDim.x * blockDim.x)
    dst[i] = f2bf(src[i]);
}

// ================= stage 1: retrieval / KL prep =================

// grid 256: rows 0..127 = text_bank rows, 128..255 = img_cls rows. block 256.
__global__ void row_stats_kernel(const float* __restrict__ text, const float* __restrict__ img,
                                 float* __restrict__ inv_t, float* __restrict__ lse_t,
                                 float* __restrict__ inv_i, float* __restrict__ Ssum,
                                 float* __restrict__ p_img) {
  __shared__ float red[256];
  int r = blockIdx.x, t = threadIdx.x;
  bool is_img = (r >= NTXT);
  const float* x = is_img ? (img + (size_t)(r - NTXT) * CDIM) : (text + (size_t)r * CDIM);
  float mx = -3e38f;
  for (int c = t; c < CDIM; c += 256) mx = fmaxf(mx, x[c]);
  red[t] = mx; __syncthreads();
  for (int s = 128; s > 0; s >>= 1) { if (t < s) red[t] = fmaxf(red[t], red[t + s]); __syncthreads(); }
  mx = red[0]; __syncthreads();
  float se = 0.f, sq = 0.f;
  for (int c = t; c < CDIM; c += 256) { float v = x[c]; se += __expf(v - mx); sq += v * v; }
  red[t] = se; __syncthreads();
  for (int s = 128; s > 0; s >>= 1) { if (t < s) red[t] += red[t + s]; __syncthreads(); }
  float lse = mx + __logf(red[0]); __syncthreads();
  red[t] = sq; __syncthreads();
  for (int s = 128; s > 0; s >>= 1) { if (t < s) red[t] += red[t + s]; __syncthreads(); }
  float inv = rsqrtf(red[0]); __syncthreads();
  if (!is_img) {
    if (t == 0) { inv_t[r] = inv; lse_t[r] = lse; }
    return;
  }
  int b = r - NTXT;
  float S = 0.f;
  for (int c = t; c < CDIM; c += 256) {
    float lp = x[c] - lse;
    float pv = __expf(lp);
    p_img[(size_t)b * CDIM + c] = pv;
    S += pv * lp;
  }
  red[t] = S; __syncthreads();
  for (int s = 128; s > 0; s >>= 1) { if (t < s) red[t] += red[t + s]; __syncthreads(); }
  if (t == 0) { inv_i[b] = inv; Ssum[b] = red[0]; }
}

// grid 128 (b), block 128 (n): sim_full[b][n]
__global__ void sim_kernel(const float* __restrict__ img_cls, const float* __restrict__ text,
                           const float* __restrict__ inv_i, const float* __restrict__ inv_t,
                           const float* __restrict__ logit_scale, float* __restrict__ sim_full) {
  int b = blockIdx.x, t = threadIdx.x;
  const float* ib = img_cls + (size_t)b * CDIM;
  const float* tb = text + (size_t)t * CDIM;
  float s = 0.f;
  for (int c = 0; c < CDIM; ++c) s += ib[c] * tb[c];
  sim_full[(size_t)b * NTXT + t] = __expf(*logit_scale) * inv_i[b] * inv_t[t] * s;
}

// grid 128 (b), block 128. Iterative argmax (ties -> lowest index), gather, init Q.
__global__ void topk_kernel(const float* __restrict__ sim_full, const float* __restrict__ text,
                            float* __restrict__ simk, int* __restrict__ idxk,
                            float* __restrict__ text_sel, float* __restrict__ Q) {
  __shared__ float vals[NTXT];
  __shared__ float rv[NTXT];
  __shared__ int   ri[NTXT];
  __shared__ int   sidx[KSEL];
  int b = blockIdx.x, t = threadIdx.x;
  vals[t] = sim_full[(size_t)b * NTXT + t];
  __syncthreads();
  for (int round = 0; round < KSEL; ++round) {
    rv[t] = vals[t]; ri[t] = t;
    __syncthreads();
    for (int s = 64; s > 0; s >>= 1) {
      if (t < s) {
        if (rv[t + s] > rv[t] || (rv[t + s] == rv[t] && ri[t + s] < ri[t])) {
          rv[t] = rv[t + s]; ri[t] = ri[t + s];
        }
      }
      __syncthreads();
    }
    if (t == 0) {
      simk[b * KSEL + round] = rv[0];
      idxk[b * KSEL + round] = ri[0];
      sidx[round] = ri[0];
      vals[ri[0]] = -3.0e38f;
    }
    __syncthreads();
  }
  for (int kk = 0; kk < KSEL; ++kk) {
    int id = sidx[kk];
    for (int c = t; c < CDIM; c += NTXT) {
      float v = text[(size_t)id * CDIM + c];
      size_t o = ((size_t)b * KSEL + kk) * CDIM + c;
      text_sel[o] = v;
      Q[o] = v;
    }
  }
}

// grid (128,16), block 256: kl[b][kk] = (S_b - <p_b, text_sel> + lse_text)/C
__global__ void kl_kernel(const float* __restrict__ p_img, const float* __restrict__ text_sel,
                          const int* __restrict__ idxk, const float* __restrict__ Ssum,
                          const float* __restrict__ lse_t, float* __restrict__ klb) {
  __shared__ float red[256];
  int b = blockIdx.x, kk = blockIdx.y, t = threadIdx.x;
  const float* p = p_img + (size_t)b * CDIM;
  const float* ts = text_sel + ((size_t)b * KSEL + kk) * CDIM;
  float d = 0.f;
  for (int c = t; c < CDIM; c += 256) d += p[c] * ts[c];
  red[t] = d; __syncthreads();
  for (int s = 128; s > 0; s >>= 1) { if (t < s) red[t] += red[t + s]; __syncthreads(); }
  if (t == 0)
    klb[b * KSEL + kk] = (Ssum[b] - red[0] + lse_t[idxk[b * KSEL + kk]]) * (1.f / CDIM);
}

// grid 128, block 32: fused softmax(sim) / softmax(-kl) weights
__global__ void weights_kernel(const float* __restrict__ simk, const float* __restrict__ klb,
                               const float* __restrict__ alpha, float* __restrict__ wfuse) {
  int b = blockIdx.x, j = threadIdx.x;
  float a = *alpha;
  float sv = (j < KSEL) ? simk[b * KSEL + j] : -3e38f;
  float kv = (j < KSEL) ? -klb[b * KSEL + j] : -3e38f;
  float m1 = sv, m2 = kv;
  for (int m = 16; m; m >>= 1) {
    m1 = fmaxf(m1, __shfl_xor(m1, m, 32));
    m2 = fmaxf(m2, __shfl_xor(m2, m, 32));
  }
  float e1 = (j < KSEL) ? __expf(sv - m1) : 0.f;
  float e2 = (j < KSEL) ? __expf(kv - m2) : 0.f;
  float s1 = e1, s2 = e2;
  for (int m = 16; m; m >>= 1) {
    s1 += __shfl_xor(s1, m, 32);
    s2 += __shfl_xor(s2, m, 32);
  }
  if (j < KSEL) wfuse[b * KSEL + j] = a * e1 / s1 + (1.f - a) * e2 / s2;
}

// ================= stage 2: transformer =================

// one wave per token: mean/invstd over contiguous 512. block 256 = 8 waves.
__global__ void ln_stats_kernel(const float* __restrict__ X, float* __restrict__ mean,
                                float* __restrict__ inv, int T) {
  int wave = threadIdx.x >> 5, lane = threadIdx.x & 31;
  int t = blockIdx.x * (blockDim.x >> 5) + wave;
  if (t >= T) return;
  const float* x = X + (size_t)t * CDIM;
  float s = 0.f, sq = 0.f;
  for (int j = lane; j < CDIM; j += 32) { float v = x[j]; s += v; sq += v * v; }
  for (int m = 16; m; m >>= 1) { s += __shfl_xor(s, m, 32); sq += __shfl_xor(sq, m, 32); }
  if (lane == 0) {
    float mu = s * (1.f / CDIM);
    mean[t] = mu;
    inv[t] = rsqrtf(sq * (1.f / CDIM) - mu * mu + EPSLN);
  }
}

// kv-token LN stats over strided img_vis columns (layer-invariant, run once)
__global__ void kv_stats_kernel(const float* __restrict__ img, float* __restrict__ mean,
                                float* __restrict__ inv) {
  int wave = threadIdx.x >> 5, lane = threadIdx.x & 31;
  long t = (long)blockIdx.x * (blockDim.x >> 5) + wave;
  if (t >= TKV) return;
  int b = (int)(t / HWNN), hw = (int)(t % HWNN);
  const float* base = img + (size_t)b * CDIM * HWNN + hw;
  float s = 0.f, sq = 0.f;
  for (int c = lane; c < CDIM; c += 32) { float v = base[(size_t)c * HWNN]; s += v; sq += v * v; }
  for (int m = 16; m; m >>= 1) { s += __shfl_xor(s, m, 32); sq += __shfl_xor(sq, m, 32); }
  if (lane == 0) {
    float mu = s * (1.f / CDIM);
    mean[t] = mu;
    inv[t] = rsqrtf(sq * (1.f / CDIM) - mu * mu + EPSLN);
  }
}

// grid-stride: out_bf16[t][c] = LN(X)[t][c] from contiguous token rows
__global__ void ln_apply_kernel(const float* __restrict__ X, const float* __restrict__ mean,
                                const float* __restrict__ inv, const float* __restrict__ g,
                                const float* __restrict__ bb, unsigned short* __restrict__ out,
                                int T) {
  size_t total = (size_t)T * CDIM;
  for (size_t i = (size_t)blockIdx.x * blockDim.x + threadIdx.x; i < total;
       i += (size_t)gridDim.x * blockDim.x) {
    size_t t = i >> 9;
    int c = (int)(i & (CDIM - 1));
    out[i] = f2bf((X[i] - mean[t]) * inv[t] * g[c] + bb[c]);
  }
}

// per-layer: kvn_bf[t][c] = LN1(img_vis token t)[c], gathered from strided (b,c,hw) layout.
// Writes coalesced bf16 rows; reads hit L2 (img_vis re-streamed).
__global__ void kvn_apply_kernel(const float* __restrict__ img, const float* __restrict__ mean,
                                 const float* __restrict__ inv, const float* __restrict__ g,
                                 const float* __restrict__ bb, unsigned short* __restrict__ out) {
  size_t total = (size_t)TKV * CDIM;
  for (size_t i = (size_t)blockIdx.x * blockDim.x + threadIdx.x; i < total;
       i += (size_t)gridDim.x * blockDim.x) {
    size_t t = i >> 9;
    int c = (int)(i & (CDIM - 1));
    int b = (int)(t / HWNN), hw = (int)(t % HWNN);
    float x = img[(size_t)b * CDIM * HWNN + (size_t)c * HWNN + hw];
    out[i] = f2bf((x - mean[t]) * inv[t] * g[c] + bb[c]);
  }
}

// Generic GEMM, all-bf16 operands: out[m][n] = ACT( A[m][:] . W[n][:] + bias[n] )
// Each wave computes a 32x64 strip: 2 A frags x 4 B frags = 8 WMMAs per k-step.
// grid (M/32, N/256), block 128 (4 waves).
template <int ACT, int STORE_BF>
__global__ void gemm_tpl(const unsigned short* __restrict__ A, int lda,
                         const unsigned short* __restrict__ Wb, int ldw,
                         const float* __restrict__ bias, int Kc,
                         unsigned short* __restrict__ outb, int ldob,
                         float* __restrict__ outacc, int ldoa) {
  int lane = threadIdx.x & 31;
  int wave = threadIdx.x >> 5;
  long row0 = (long)blockIdx.x * 32;
  long col0 = (long)blockIdx.y * 256 + wave * 64;
  v8f acc[2][4];
#pragma unroll
  for (int mi = 0; mi < 2; ++mi)
#pragma unroll
    for (int nt = 0; nt < 4; ++nt)
      acc[mi][nt] = (v8f){0.f, 0.f, 0.f, 0.f, 0.f, 0.f, 0.f, 0.f};
  FragBF fa0, fa1, fb;
  const unsigned short* arow = A + (size_t)(row0 + (lane & 15)) * lda;
  for (int kb = 0; kb < Kc; kb += 32) {
    loadA_bf(fa0, A, lda, row0, kb, lane);
    loadA_bf(fa1, A, lda, row0 + 16, kb, lane);
    if (kb + 32 < Kc) __builtin_prefetch(arow + kb + 32, 0, 0);   // global_prefetch_b8
#pragma unroll
    for (int nt = 0; nt < 4; ++nt) {
      loadB_nk_bf(fb, Wb, ldw, col0 + nt * 16, kb, lane);
      acc[0][nt] = wmma_bf16(fa0, fb, acc[0][nt]);
      acc[1][nt] = wmma_bf16(fa1, fb, acc[1][nt]);
    }
  }
  int n = lane & 15, mh = lane >> 4;
#pragma unroll
  for (int mi = 0; mi < 2; ++mi) {
#pragma unroll
    for (int nt = 0; nt < 4; ++nt) {
      long cc = col0 + nt * 16 + n;
      float bn = bias[cc];
#pragma unroll
      for (int r = 0; r < 8; ++r) {
        float vv = acc[mi][nt][r] + bn;
        if (ACT) vv = quick_gelu(vv);
        long m = row0 + mi * 16 + mh * 8 + r;
        if (STORE_BF) outb[(size_t)m * ldob + cc] = f2bf(vv);
        else          outacc[(size_t)m * ldoa + cc] += vv;
      }
    }
  }
}

// Cross-attention per (b, h): scores WMMA -> LDS softmax (writes bf16 probs) -> attn@V WMMA.
// grid (128, 8), block 128 (4 waves). LDS = 36 KB f32 + 18 KB bf16.
__global__ void attn_kernel(const unsigned short* __restrict__ Qp,
                            const unsigned short* __restrict__ KVp,
                            unsigned short* __restrict__ Obf) {
  __shared__ float sc[16 * HWNN];
  __shared__ unsigned short scb[16 * HWNN];
  int b = blockIdx.x, h = blockIdx.y;
  int lane = threadIdx.x & 31, wave = threadIdx.x >> 5;
  long qrow0 = (long)b * KSEL;
  long krow0 = (long)b * HWNN;
  FragBF fa, fb;
  // phase 1: scores = (Qh . Kh^T) / 8
  for (int ct = wave; ct < HWNN / 16; ct += 4) {
    v8f acc = {0.f, 0.f, 0.f, 0.f, 0.f, 0.f, 0.f, 0.f};
    for (int kc = 0; kc < DHEAD; kc += 32) {
      loadA_bf(fa, Qp, CDIM, qrow0, h * DHEAD + kc, lane);
      loadB_nk_bf(fb, KVp, 1024, krow0 + ct * 16, h * DHEAD + kc, lane);
      acc = wmma_bf16(fa, fb, acc);
    }
    int n = lane & 15, mh = lane >> 4;
#pragma unroll
    for (int r = 0; r < 8; ++r)
      sc[(mh * 8 + r) * HWNN + ct * 16 + n] = acc[r] * 0.125f;
  }
  __syncthreads();
  // phase 2: row softmax over 576; emit normalized probs as bf16 tile
  {
    int g = threadIdx.x >> 4;   // 0..7
    int j = threadIdx.x & 15;
    for (int rr = 0; rr < 2; ++rr) {
      int row = g + rr * 8;
      float* p = sc + (size_t)row * HWNN;
      float mx = -3e38f;
      for (int c = j; c < HWNN; c += 16) mx = fmaxf(mx, p[c]);
      for (int m = 8; m; m >>= 1) mx = fmaxf(mx, __shfl_xor(mx, m, 16));
      float sum = 0.f;
      for (int c = j; c < HWNN; c += 16) { float e = __expf(p[c] - mx); p[c] = e; sum += e; }
      for (int m = 8; m; m >>= 1) sum += __shfl_xor(sum, m, 16);
      float is = __builtin_amdgcn_rcpf(sum);
      for (int c = j; c < HWNN; c += 16) scb[(size_t)row * HWNN + c] = f2bf(p[c] * is);
    }
  }
  __syncthreads();
  // phase 3: O = attn @ Vh  (wave -> one 16-wide d tile, bf16 LDS A operand)
  {
    int dt = wave;
    v8f acc = {0.f, 0.f, 0.f, 0.f, 0.f, 0.f, 0.f, 0.f};
    for (int kc = 0; kc < HWNN; kc += 32) {
      loadA_bf(fa, scb, HWNN, 0, kc, lane);
      loadB_kn_bf(fb, KVp, 1024, krow0 + kc, CDIM + h * DHEAD + dt * 16, lane);
      acc = wmma_bf16(fa, fb, acc);
    }
    int n = lane & 15, mh = lane >> 4;
#pragma unroll
    for (int r = 0; r < 8; ++r)
      Obf[(size_t)(qrow0 + mh * 8 + r) * CDIM + h * DHEAD + dt * 16 + n] = f2bf(acc[r]);
  }
}

// ================= stage 3: fusion + outputs =================

// grid 2048 (b*16+kk), block 256: final LN + fused weighting -> text_out (+bf16 copy)
__global__ void fusion_kernel(const float* __restrict__ Q, const float* __restrict__ qmean,
                              const float* __restrict__ qinv, const float* __restrict__ gf,
                              const float* __restrict__ bf_, const float* __restrict__ text_sel,
                              const float* __restrict__ wfuse,
                              float* __restrict__ out_text, unsigned short* __restrict__ tob) {
  int t = blockIdx.x;
  float wv = wfuse[t];
  float mu = qmean[t], iv = qinv[t];
  for (int c = threadIdx.x; c < CDIM; c += blockDim.x) {
    float qv = Q[(size_t)t * CDIM + c];
    float cross = (qv - mu) * iv * gf[c] + bf_[c];
    float ts = text_sel[(size_t)t * CDIM + c];
    float to = wv * ts + cross + ts;
    out_text[(size_t)t * CDIM + c] = to;
    tob[(size_t)t * CDIM + c] = f2bf(to);
  }
}

// copy img_vis into channel block [0,512) of output
__global__ void copy_img_kernel(const float* __restrict__ img, float* __restrict__ out) {
  size_t total = (size_t)BATCH * CDIM * HWNN;
  for (size_t i = (size_t)blockIdx.x * blockDim.x + threadIdx.x; i < total;
       i += (size_t)gridDim.x * blockDim.x) {
    size_t b = i / ((size_t)CDIM * HWNN);
    size_t rem = i - b * (size_t)CDIM * HWNN;
    out[b * (size_t)(CDIM + KSEL) * HWNN + rem] = img[i];
  }
}

// score_map[b][kk][hw] = text_out[b][kk][:] . img_vis[b][:][hw] via WMMA
// grid (128, 9), block 128.
__global__ void score_kernel(const unsigned short* __restrict__ tob,
                             const float* __restrict__ img, float* __restrict__ out) {
  int b = blockIdx.x;
  int nt = blockIdx.y * 4 + (threadIdx.x >> 5);   // 0..35 column tile over hw
  int lane = threadIdx.x & 31;
  v8f acc = {0.f, 0.f, 0.f, 0.f, 0.f, 0.f, 0.f, 0.f};
  FragBF fa, fb;
  const float* imgb = img + (size_t)b * CDIM * HWNN;
  for (int kc = 0; kc < CDIM; kc += 32) {
    loadA_bf(fa, tob, CDIM, (long)b * KSEL, kc, lane);
    loadB_kn_f32(fb, imgb, HWNN, kc, nt * 16, lane);
    acc = wmma_bf16(fa, fb, acc);
  }
  int n = lane & 15, mh = lane >> 4;
#pragma unroll
  for (int r = 0; r < 8; ++r)
    out[(size_t)b * ((CDIM + KSEL) * HWNN) + (size_t)(CDIM + mh * 8 + r) * HWNN + nt * 16 + n] =
        acc[r];
}

// ================= host orchestration =================

extern "C" void kernel_launch(void* const* d_in, const int* in_sizes, int n_in,
                              void* d_out, int out_size, void* d_ws, size_t ws_size,
                              hipStream_t stream) {
  (void)in_sizes; (void)n_in; (void)out_size; (void)ws_size;
  const float* img_vis     = (const float*)d_in[0];
  const float* img_cls     = (const float*)d_in[1];
  const float* text_bank   = (const float*)d_in[2];
  const float* logit_scale = (const float*)d_in[3];
  const float* alpha       = (const float*)d_in[4];
  const float* in_proj_w   = (const float*)d_in[5];
  const float* in_proj_b   = (const float*)d_in[6];
  const float* out_proj_w  = (const float*)d_in[7];
  const float* out_proj_b  = (const float*)d_in[8];
  const float* ln1_g       = (const float*)d_in[9];
  const float* ln1_b       = (const float*)d_in[10];
  const float* fc_w        = (const float*)d_in[11];
  const float* fc_b        = (const float*)d_in[12];
  const float* proj_w      = (const float*)d_in[13];
  const float* proj_b      = (const float*)d_in[14];
  const float* ln2_g       = (const float*)d_in[15];
  const float* ln2_b       = (const float*)d_in[16];
  const float* normf_g     = (const float*)d_in[17];
  const float* normf_b     = (const float*)d_in[18];
  // d_in[19] = k (fixed to 16)

  float* out = (float*)d_out;
  const size_t OUT2 = (size_t)BATCH * (CDIM + KSEL) * HWNN;   // text_out offset

  // workspace carve (~277 MB total: KVp 151 MB + kvn 75.5 MB + weights 25 MB + rest)
  char* w = (char*)d_ws;
  auto carve = [&](size_t bytes) -> char* {
    char* p = w;
    w += (bytes + 255) & ~(size_t)255;
    return p;
  };
  float* p_img    = (float*)carve((size_t)BATCH * CDIM * 4);
  float* inv_t    = (float*)carve(NTXT * 4);
  float* lse_t    = (float*)carve(NTXT * 4);
  float* inv_i    = (float*)carve(BATCH * 4);
  float* Ssum     = (float*)carve(BATCH * 4);
  float* sim_full = (float*)carve((size_t)BATCH * NTXT * 4);
  float* simk     = (float*)carve((size_t)BATCH * KSEL * 4);
  int*   idxk     = (int*)  carve((size_t)BATCH * KSEL * 4);
  float* klb      = (float*)carve((size_t)BATCH * KSEL * 4);
  float* wfuse    = (float*)carve((size_t)BATCH * KSEL * 4);
  float* text_sel = (float*)carve((size_t)TQ * CDIM * 4);
  float* Qbuf     = (float*)carve((size_t)TQ * CDIM * 4);
  float* qmean    = (float*)carve((size_t)TQ * 4);
  float* qinv     = (float*)carve((size_t)TQ * 4);
  float* kvmean   = (float*)carve((size_t)TKV * 4);
  float* kvinv    = (float*)carve((size_t)TKV * 4);
  unsigned short* qn_bf  = (unsigned short*)carve((size_t)TQ * CDIM * 2);
  unsigned short* Qp_bf  = (unsigned short*)carve((size_t)TQ * CDIM * 2);
  unsigned short* O_bf   = (unsigned short*)carve((size_t)TQ * CDIM * 2);
  unsigned short* H1_bf  = (unsigned short*)carve((size_t)TQ * FDIM * 2);
  unsigned short* tob_bf = (unsigned short*)carve((size_t)TQ * CDIM * 2);
  // bf16 weight images (converted once per launch)
  const size_t IN_W  = (size_t)NLAYER * 3 * CDIM * CDIM;
  const size_t OUT_W = (size_t)NLAYER * CDIM * CDIM;
  const size_t FC_W  = (size_t)NLAYER * FDIM * CDIM;
  const size_t PR_W  = (size_t)NLAYER * CDIM * FDIM;
  unsigned short* inW_bf   = (unsigned short*)carve(IN_W * 2);
  unsigned short* outW_bf  = (unsigned short*)carve(OUT_W * 2);
  unsigned short* fcW_bf   = (unsigned short*)carve(FC_W * 2);
  unsigned short* prW_bf   = (unsigned short*)carve(PR_W * 2);
  unsigned short* kvn_bf   = (unsigned short*)carve((size_t)TKV * CDIM * 2);
  unsigned short* KVp      = (unsigned short*)carve((size_t)TKV * 1024 * 2);

  // ---- weight preconversion to bf16 (keeps GEMM hot loops load+wmma only) ----
  cvt_bf16_kernel<<<2048, 256, 0, stream>>>(in_proj_w, inW_bf, IN_W);
  cvt_bf16_kernel<<<2048, 256, 0, stream>>>(out_proj_w, outW_bf, OUT_W);
  cvt_bf16_kernel<<<2048, 256, 0, stream>>>(fc_w, fcW_bf, FC_W);
  cvt_bf16_kernel<<<2048, 256, 0, stream>>>(proj_w, prW_bf, PR_W);

  // ---- retrieval / KL weighting ----
  row_stats_kernel<<<256, 256, 0, stream>>>(text_bank, img_cls, inv_t, lse_t, inv_i, Ssum, p_img);
  sim_kernel<<<BATCH, NTXT, 0, stream>>>(img_cls, text_bank, inv_i, inv_t, logit_scale, sim_full);
  topk_kernel<<<BATCH, NTXT, 0, stream>>>(sim_full, text_bank, simk, idxk, text_sel, Qbuf);
  kl_kernel<<<dim3(BATCH, KSEL), 256, 0, stream>>>(p_img, text_sel, idxk, Ssum, lse_t, klb);
  weights_kernel<<<BATCH, 32, 0, stream>>>(simk, klb, alpha, wfuse);

  // kv-token LN stats (layer invariant)
  kv_stats_kernel<<<TKV / 8, 256, 0, stream>>>(img_vis, kvmean, kvinv);

  // ---- 4-layer cross transformer ----
  for (int l = 0; l < NLAYER; ++l) {
    const unsigned short* Wl = inW_bf + (size_t)l * 3 * CDIM * CDIM;
    const float*          Bl = in_proj_b + (size_t)l * 3 * CDIM;
    const unsigned short* Wo = outW_bf + (size_t)l * CDIM * CDIM;
    const float*          Bo = out_proj_b + (size_t)l * CDIM;
    const float*          g1 = ln1_g + (size_t)l * CDIM;
    const float*          b1 = ln1_b + (size_t)l * CDIM;
    const unsigned short* Wf = fcW_bf + (size_t)l * FDIM * CDIM;
    const float*          Bf = fc_b + (size_t)l * FDIM;
    const unsigned short* Wp = prW_bf + (size_t)l * CDIM * FDIM;
    const float*          Bp = proj_b + (size_t)l * CDIM;
    const float*          g2 = ln2_g + (size_t)l * CDIM;
    const float*          b2 = ln2_b + (size_t)l * CDIM;

    // qn = LN1(Q) -> bf16 ; Qp = qn @ Wq^T + bq
    ln_stats_kernel<<<TQ / 8, 256, 0, stream>>>(Qbuf, qmean, qinv, TQ);
    ln_apply_kernel<<<2048, 256, 0, stream>>>(Qbuf, qmean, qinv, g1, b1, qn_bf, TQ);
    gemm_tpl<0, 1><<<dim3(TQ / 32, CDIM / 256), 128, 0, stream>>>(
        qn_bf, CDIM, Wl, CDIM, Bl, CDIM, Qp_bf, CDIM, nullptr, 0);
    // kvn = LN1(img tokens) -> bf16 (single bandwidth pass, then pure-WMMA GEMM)
    kvn_apply_kernel<<<8192, 256, 0, stream>>>(img_vis, kvmean, kvinv, g1, b1, kvn_bf);
    // K/V projection (dominant GEMM: 73728 x 1024 x 512), all-bf16 operands
    gemm_tpl<0, 1><<<dim3(TKV / 32, 1024 / 256), 128, 0, stream>>>(
        kvn_bf, CDIM, Wl + (size_t)CDIM * CDIM, CDIM, Bl + CDIM, CDIM, KVp, 1024, nullptr, 0);
    // attention
    attn_kernel<<<dim3(BATCH, NHEAD), 128, 0, stream>>>(Qp_bf, KVp, O_bf);
    // out proj + residual into Q
    gemm_tpl<0, 0><<<dim3(TQ / 32, CDIM / 256), 128, 0, stream>>>(
        O_bf, CDIM, Wo, CDIM, Bo, CDIM, nullptr, 0, Qbuf, CDIM);
    // MLP: h = quickGELU(LN2(Q) @ fc^T + b) @ proj^T + b, residual into Q
    ln_stats_kernel<<<TQ / 8, 256, 0, stream>>>(Qbuf, qmean, qinv, TQ);
    ln_apply_kernel<<<2048, 256, 0, stream>>>(Qbuf, qmean, qinv, g2, b2, qn_bf, TQ);
    gemm_tpl<1, 1><<<dim3(TQ / 32, FDIM / 256), 128, 0, stream>>>(
        qn_bf, CDIM, Wf, CDIM, Bf, CDIM, H1_bf, FDIM, nullptr, 0);
    gemm_tpl<0, 0><<<dim3(TQ / 32, CDIM / 256), 128, 0, stream>>>(
        H1_bf, FDIM, Wp, FDIM, Bp, FDIM, nullptr, 0, Qbuf, CDIM);
  }

  // ---- final LN + fusion + outputs ----
  ln_stats_kernel<<<TQ / 8, 256, 0, stream>>>(Qbuf, qmean, qinv, TQ);
  fusion_kernel<<<TQ, 256, 0, stream>>>(Qbuf, qmean, qinv, normf_g, normf_b, text_sel, wfuse,
                                        out + OUT2, tob_bf);
  copy_img_kernel<<<4096, 256, 0, stream>>>(img_vis, out);
  score_kernel<<<dim3(BATCH, (HWNN / 16) / 4), 128, 0, stream>>>(tob_bf, img_vis, out);
}
